// ConAttn_59897613910138
// MI455X (gfx1250) — matrix operand order repaired and checked by reading, not compiled, not run on hardware
//
#include <hip/hip_runtime.h>
#include <hip/hip_bf16.h>
#include <math.h>

typedef _Float16 v16h __attribute__((ext_vector_type(16)));
typedef float    v8f  __attribute__((ext_vector_type(8)));

#define C_DIM 128
#define N_TOK 4096
#define NHEAD 4
#define DHEAD 32
#define BATCH 2

__device__ __forceinline__ float leakyf(float t) {
    return t >= 0.f ? t : 0.2f * t;
}

// ---------------------------------------------------------------------------
// Kernel 1: per-token 1x1 convs (q_feat, v_feat), k normalization, gate MLPs.
// grid = B*N blocks, 128 threads (thread t = output channel).
// ---------------------------------------------------------------------------
__global__ void __launch_bounds__(128)
proj_kernel(const float* __restrict__ x,
            const float* __restrict__ Wq, const float* __restrict__ bq,
            const float* __restrict__ Wv, const float* __restrict__ bv,
            const float* __restrict__ lw1, const float* __restrict__ lb1,
            const float* __restrict__ lw2, const float* __restrict__ lb2,
            const float* __restrict__ bw1, const float* __restrict__ bb1,
            const float* __restrict__ bw2, const float* __restrict__ bb2,
            _Float16* __restrict__ qh, _Float16* __restrict__ kh,
            _Float16* __restrict__ vh, float* __restrict__ wgt,
            float* __restrict__ bia) {
    __shared__ float sx[C_DIM];
    __shared__ float sq[C_DIM];
    __shared__ float red[C_DIM];
    __shared__ float sh1[DHEAD];
    __shared__ float sh2[DHEAD];

    const int n = blockIdx.x & (N_TOK - 1);
    const int b = blockIdx.x >> 12;
    const int t = threadIdx.x;

    sx[t] = x[((size_t)(b * C_DIM + t)) * N_TOK + n];
    __syncthreads();

    float q = bq[t], v = bv[t];
    const float* wqr = Wq + (size_t)t * C_DIM;
    const float* wvr = Wv + (size_t)t * C_DIM;
    #pragma unroll 4
    for (int c = 0; c < C_DIM; ++c) {
        q = fmaf(wqr[c], sx[c], q);
        v = fmaf(wvr[c], sx[c], v);
    }
    sq[t]  = q;
    red[t] = q * q;
    __syncthreads();
    #pragma unroll
    for (int s = 64; s > 0; s >>= 1) {
        if (t < s) red[t] += red[t + s];
        __syncthreads();
    }
    const float nrm = fmaxf(sqrtf(red[0]), 1e-4f);
    const size_t tok = (size_t)b * N_TOK + n;
    qh[tok * C_DIM + t] = (_Float16)q;
    kh[tok * C_DIM + t] = (_Float16)(q / nrm);
    vh[tok * C_DIM + t] = (_Float16)v;

    // gate MLPs on q_feat (pre-normalization)
    if (t < DHEAD) {
        float h1 = lb1[t], h2 = bb1[t];
        const float* l1 = lw1 + (size_t)t * C_DIM;
        const float* b1 = bw1 + (size_t)t * C_DIM;
        #pragma unroll 4
        for (int c = 0; c < C_DIM; ++c) {
            h1 = fmaf(l1[c], sq[c], h1);
            h2 = fmaf(b1[c], sq[c], h2);
        }
        sh1[t] = leakyf(h1);
        sh2[t] = leakyf(h2);
    }
    __syncthreads();
    if (t == 0) {
        float w = lb2[0], bb = bb2[0];
        #pragma unroll
        for (int c = 0; c < DHEAD; ++c) {
            w  = fmaf(lw2[c], sh1[c], w);
            bb = fmaf(bw2[c], sh2[c], bb);
        }
        wgt[tok] = w;
        bia[tok] = bb;
    }
}

// ---------------------------------------------------------------------------
// Kernel 2: flash attention with dual accumulators (y, yw) via WMMA f16.
// Fixed softmax bound m_r = |q_row| (valid: k is unit-norm so every score
// <= |q_row|) removes all per-tile max/rescale work; row-sum accumulated by
// an extra WMMA against a ones matrix (no shuffle reductions in the loop).
// grid = B*NHEAD*(N/128) = 256 blocks, 256 threads (8 waves, 16 rows/wave).
// ---------------------------------------------------------------------------
__global__ void __launch_bounds__(256)
attn_kernel(const _Float16* __restrict__ qh, const _Float16* __restrict__ kh,
            const _Float16* __restrict__ vh, const float* __restrict__ wgt,
            float* __restrict__ y_ws, float* __restrict__ yw_ws) {
    __shared__ _Float16 sK [32][DHEAD];         // 32 keys x dq
    __shared__ _Float16 sV [32][DHEAD];         // 32 keys x dv
    __shared__ _Float16 sVW[32][DHEAD];         // gate-scaled V
    __shared__ _Float16 sP [8][16][32];         // per-wave P staging

    const int lane = threadIdx.x & 31;
    const int wave = threadIdx.x >> 5;
    const int m    = lane & 15;
    const int hi   = lane >> 4;

    const int idx = blockIdx.x;
    const int rb  = idx & 31;
    const int h   = (idx >> 5) & 3;
    const int b   = idx >> 7;

    // --- resident Q fragment (A-matrix, 16x32 f16 layout) ---
    v16h aq;
    {
        const _Float16* qrow =
            qh + ((size_t)b * N_TOK + rb * 128 + wave * 16 + m) * C_DIM + h * DHEAD;
        #pragma unroll
        for (int j = 0; j < 16; ++j) {
            const int kk = j + (hi ? 8 : 0) + (j >= 8 ? 8 : 0);
            aq[j] = qrow[kk];
        }
    }

    // --- fixed per-row softmax shift: m_r = |q_head row| + eps ---
    float mrow[8];
    {
        float ssq = 0.f;
        #pragma unroll
        for (int j = 0; j < 16; ++j) {
            const float f = (float)aq[j];
            ssq = fmaf(f, f, ssq);
        }
        ssq += __shfl_xor(ssq, 16);            // join the two K-halves of row m
        const float tot = sqrtf(ssq) + 1e-3f;  // slack for f16 rounding
        #pragma unroll
        for (int r = 0; r < 8; ++r)
            mrow[r] = __shfl(tot, hi ? (24 + r) : r);  // lane holding row hi*8+r
    }

    v16h ones;
    #pragma unroll
    for (int j = 0; j < 16; ++j) ones[j] = (_Float16)1.0f;

    v8f acc_y0 = {}, acc_y1 = {}, acc_w0 = {}, acc_w1 = {}, acc_l = {};

    const int NT = N_TOK / 32;
    for (int kt = 0; kt < NT; ++kt) {
        __syncthreads();
        // cooperative K/V tile load; gate-scaled V computed once per block
        {
            const int t = threadIdx.x;
            const size_t kbase = ((size_t)b * N_TOK + (size_t)kt * 32);
            #pragma unroll
            for (int i = 0; i < 4; ++i) {
                const int e   = t + i * 256;
                const int key = e >> 5, d = e & 31;
                const float w = wgt[kbase + key];
                sK[key][d] = kh[(kbase + key) * C_DIM + h * DHEAD + d];
                const float vv = (float)vh[(kbase + key) * C_DIM + h * DHEAD + d];
                sV [key][d] = (_Float16)vv;
                sVW[key][d] = (_Float16)(vv * w);
            }
            if (kt + 1 < NT && t < 32) {
                const size_t nb = ((size_t)b * N_TOK + (size_t)(kt + 1) * 32 + t) * C_DIM + h * DHEAD;
                __builtin_prefetch(kh + nb, 0, 0);   // global_prefetch_b8
                __builtin_prefetch(vh + nb, 0, 0);
            }
        }
        __syncthreads();

        // --- scores: S[16 rows x 32 keys] = Q(16x32) @ K^T ---
        v16h bk0, bk1;
        #pragma unroll
        for (int j = 0; j < 16; ++j) {
            const int kk = j + (hi ? 8 : 0) + (j >= 8 ? 8 : 0);  // dq index (B-matrix K)
            bk0[j] = sK[m][kk];                        // key column m
            bk1[j] = sK[m + 16][kk];                   // key column m+16
        }
        const v8f zero = {};
        v8f s0 = __builtin_amdgcn_wmma_f32_16x16x32_f16(false, aq, false, bk0,
                                                        (short)0, zero, false, false);
        v8f s1 = __builtin_amdgcn_wmma_f32_16x16x32_f16(false, aq, false, bk1,
                                                        (short)0, zero, false, false);

        // --- p = exp(s - m_fixed): no reductions, no rescaling ---
        #pragma unroll
        for (int r = 0; r < 8; ++r) {
            const float p0 = __expf(s0[r] - mrow[r]);
            const float p1 = __expf(s1[r] - mrow[r]);
            const int rr = hi * 8 + r;                 // C-layout row
            sP[wave][rr][m]      = (_Float16)p0;
            sP[wave][rr][m + 16] = (_Float16)p1;
        }

        // --- reload P as A fragment (same-wave LDS, in-order DS pipe) ---
        v16h ap;
        #pragma unroll
        for (int j = 0; j < 16; ++j) {
            const int kk = j + (hi ? 8 : 0) + (j >= 8 ? 8 : 0);  // key index
            ap[j] = sP[wave][m][kk];
        }

        // --- row-sum of P via WMMA against ones(32x16) ---
        acc_l = __builtin_amdgcn_wmma_f32_16x16x32_f16(false, ap, false, ones,
                                                       (short)0, acc_l, false, false);

        // --- V and gated-V B fragments, dv split into two 16-col halves ---
        v16h bv0, bv1, bw0, bw1;
        #pragma unroll
        for (int j = 0; j < 16; ++j) {
            const int key = j + (hi ? 8 : 0) + (j >= 8 ? 8 : 0);  // B-matrix K = key
            bv0[j] = sV [key][m];
            bv1[j] = sV [key][m + 16];
            bw0[j] = sVW[key][m];
            bw1[j] = sVW[key][m + 16];
        }
        acc_y0 = __builtin_amdgcn_wmma_f32_16x16x32_f16(false, ap, false, bv0,
                                                        (short)0, acc_y0, false, false);
        acc_y1 = __builtin_amdgcn_wmma_f32_16x16x32_f16(false, ap, false, bv1,
                                                        (short)0, acc_y1, false, false);
        acc_w0 = __builtin_amdgcn_wmma_f32_16x16x32_f16(false, ap, false, bw0,
                                                        (short)0, acc_w0, false, false);
        acc_w1 = __builtin_amdgcn_wmma_f32_16x16x32_f16(false, ap, false, bw1,
                                                        (short)0, acc_w1, false, false);
    }

    // --- normalize and write y / yw (acc_l holds row-sum in every column) ---
    #pragma unroll
    for (int r = 0; r < 8; ++r) {
        const float inv = 1.0f / acc_l[r];
        const int   rr  = hi * 8 + r;
        const int orow  = rb * 128 + wave * 16 + rr;
        const size_t base = (((size_t)(b * NHEAD + h) * N_TOK) + orow) * DHEAD;
        y_ws [base + m]      = acc_y0[r] * inv;
        y_ws [base + 16 + m] = acc_y1[r] * inv;
        yw_ws[base + m]      = acc_w0[r] * inv;
        yw_ws[base + 16 + m] = acc_w1[r] * inv;
    }
}

// ---------------------------------------------------------------------------
// Kernel 3: background = mean_n(yw), bias_value = sum_n(bia*v). grid = B*h.
// ---------------------------------------------------------------------------
__global__ void __launch_bounds__(256)
reduce_kernel(const float* __restrict__ yw_ws, const float* __restrict__ bia,
              const _Float16* __restrict__ vh,
              float* __restrict__ bg, float* __restrict__ bvv) {
    const int bh = blockIdx.x;          // 0..7
    const int b  = bh >> 2, h = bh & 3;
    const int t  = threadIdx.x;
    const int d  = t & 31;
    const int ch = t >> 5;              // 0..7
    float s1 = 0.f, s2 = 0.f;
    for (int n = ch; n < N_TOK; n += 8) {
        s1 += yw_ws[(((size_t)bh * N_TOK) + n) * DHEAD + d];
        s2 += bia[(size_t)b * N_TOK + n] *
              (float)vh[((size_t)b * N_TOK + n) * C_DIM + h * DHEAD + d];
    }
    __shared__ float r1[8][32], r2[8][32];
    r1[ch][d] = s1; r2[ch][d] = s2;
    __syncthreads();
    if (ch == 0) {
        float a = 0.f, c = 0.f;
        #pragma unroll
        for (int i = 0; i < 8; ++i) { a += r1[i][d]; c += r2[i][d]; }
        bg [bh * DHEAD + d] = a / (float)N_TOK;
        bvv[bh * DHEAD + d] = c;
    }
}

// ---------------------------------------------------------------------------
// Kernel 4: out = y + relu(lam)*relu(yw - bg + bvv), scatter to NCHW features.
// ---------------------------------------------------------------------------
__global__ void __launch_bounds__(256)
combine_kernel(const float* __restrict__ y_ws, const float* __restrict__ yw_ws,
               const float* __restrict__ bg, const float* __restrict__ bvv,
               const float* __restrict__ lam, float* __restrict__ out_tok) {
    const size_t i = (size_t)blockIdx.x * 256 + threadIdx.x;  // B*h*N*32
    const int d  = (int)(i & 31);
    const size_t r = i >> 5;
    const int n  = (int)(r & (N_TOK - 1));
    const int bh = (int)(r >> 12);
    const int h = bh & 3, b = bh >> 2;
    const float lamr = fmaxf(lam[0], 0.f);
    const float con  = yw_ws[i] - bg[bh * DHEAD + d] + bvv[bh * DHEAD + d];
    const float o    = y_ws[i] + lamr * fmaxf(con, 0.f);
    out_tok[((size_t)(b * C_DIM + h * DHEAD + d)) * N_TOK + n] = o;
}

// ---------------------------------------------------------------------------
// Kernel 5: 3x3 SAME conv + bias + leaky + residual. one thread per output.
// ---------------------------------------------------------------------------
__global__ void __launch_bounds__(256)
conv3_kernel(const float* __restrict__ ft, const float* __restrict__ Wout,
             const float* __restrict__ bout, const float* __restrict__ xin,
             float* __restrict__ out) {
    const size_t i = (size_t)blockIdx.x * 256 + threadIdx.x;  // B*128*4096
    const int p  = (int)(i & (N_TOK - 1));
    const int co = (int)((i >> 12) & 127);
    const int b  = (int)(i >> 19);
    const int yy = p >> 6, xx = p & 63;
    float acc = bout[co];
    for (int ci = 0; ci < C_DIM; ++ci) {
        const float* fp = ft + ((size_t)(b * C_DIM + ci)) * N_TOK;
        const float* wp = Wout + ((size_t)(co * C_DIM + ci)) * 9;
        #pragma unroll
        for (int ky = 0; ky < 3; ++ky) {
            const int iy = yy + ky - 1;
            if (iy < 0 || iy > 63) continue;
            #pragma unroll
            for (int kx = 0; kx < 3; ++kx) {
                const int ix = xx + kx - 1;
                if (ix < 0 || ix > 63) continue;
                acc = fmaf(wp[ky * 3 + kx], fp[iy * 64 + ix], acc);
            }
        }
    }
    out[i] = leakyf(acc) + xin[i];
}

// ---------------------------------------------------------------------------
extern "C" void kernel_launch(void* const* d_in, const int* in_sizes, int n_in,
                              void* d_out, int out_size, void* d_ws, size_t ws_size,
                              hipStream_t stream) {
    (void)in_sizes; (void)n_in; (void)out_size; (void)ws_size;
    const float* x    = (const float*)d_in[0];
    const float* Wq   = (const float*)d_in[1];
    const float* bq   = (const float*)d_in[2];
    const float* Wv   = (const float*)d_in[3];
    const float* bv   = (const float*)d_in[4];
    const float* lw1  = (const float*)d_in[5];
    const float* lb1  = (const float*)d_in[6];
    const float* lw2  = (const float*)d_in[7];
    const float* lb2  = (const float*)d_in[8];
    const float* bw1  = (const float*)d_in[9];
    const float* bb1  = (const float*)d_in[10];
    const float* bw2  = (const float*)d_in[11];
    const float* bb2  = (const float*)d_in[12];
    const float* lam  = (const float*)d_in[13];
    const float* Wout = (const float*)d_in[14];
    const float* bout = (const float*)d_in[15];
    float* out = (float*)d_out;

    char* ws = (char*)d_ws;
    const size_t tokC = (size_t)BATCH * N_TOK * C_DIM;          // 1,048,576
    _Float16* qh  = (_Float16*)ws; ws += tokC * sizeof(_Float16);
    _Float16* kh  = (_Float16*)ws; ws += tokC * sizeof(_Float16);
    _Float16* vh  = (_Float16*)ws; ws += tokC * sizeof(_Float16);
    float* wgt    = (float*)ws;    ws += (size_t)BATCH * N_TOK * sizeof(float);
    float* bia    = (float*)ws;    ws += (size_t)BATCH * N_TOK * sizeof(float);
    float* y_ws   = (float*)ws;    ws += tokC * sizeof(float);
    float* yw_ws  = (float*)ws;    ws += tokC * sizeof(float);
    float* bg     = (float*)ws;    ws += (size_t)BATCH * NHEAD * DHEAD * sizeof(float);
    float* bvv    = (float*)ws;    ws += (size_t)BATCH * NHEAD * DHEAD * sizeof(float);
    float* outtok = (float*)ws;    ws += tokC * sizeof(float);

    proj_kernel<<<BATCH * N_TOK, 128, 0, stream>>>(
        x, Wq, bq, Wv, bv, lw1, lb1, lw2, lb2, bw1, bb1, bw2, bb2,
        qh, kh, vh, wgt, bia);

    attn_kernel<<<BATCH * NHEAD * (N_TOK / 128), 256, 0, stream>>>(
        qh, kh, vh, wgt, y_ws, yw_ws);

    reduce_kernel<<<BATCH * NHEAD, 256, 0, stream>>>(yw_ws, bia, vh, bg, bvv);

    combine_kernel<<<(unsigned)(tokC / 256), 256, 0, stream>>>(
        y_ws, yw_ws, bg, bvv, lam, outtok);

    conv3_kernel<<<(unsigned)(tokC / 256), 256, 0, stream>>>(
        outtok, Wout, bout, x, out);
}
